// Decay2DBlk_29094108463322
// MI455X (gfx1250) — matrix operand order repaired and checked by reading, not compiled, not run on hardware
//
#include <hip/hip_runtime.h>
#include <hip/hip_bf16.h>
#include <math.h>

typedef __bf16 bf16;
typedef __attribute__((ext_vector_type(16))) __bf16 v16bf;
typedef __attribute__((ext_vector_type(8)))  float  v8f;

#define DECAYF     0.99f
#define LOG_DECAY  (-0.0100503358535014405f)

__device__ __forceinline__ v8f wmma_bf16(v16bf a, v16bf b, v8f c) {
  return __builtin_amdgcn_wmma_f32_16x16x32_bf16(false, a, false, b, (short)0, c, false, false);
}

// ---------------- TDM (Tensor Data Mover) support ----------------
#if __has_builtin(__builtin_amdgcn_tensor_load_to_lds) && __has_builtin(__builtin_amdgcn_s_wait_tensorcnt)
#define HAS_TDM 1
typedef unsigned int tdm_u32x4 __attribute__((ext_vector_type(4)));
typedef int          tdm_i32x8 __attribute__((ext_vector_type(8)));
typedef int          tdm_i32x4 __attribute__((ext_vector_type(4)));

// 2D bf16 tile load: tile (tile0 x tile1) from tensor (dim0 x dim1), row stride
// stride0 (elements). Descriptor bitfields per CDNA5 ISA.
__device__ __forceinline__ void tdm_load_tile_2d(unsigned lds_addr, const void* gsrc,
                                                 unsigned tile0, unsigned tile1,
                                                 unsigned dim0, unsigned dim1,
                                                 unsigned long long stride0) {
  unsigned long long ga = (unsigned long long)(size_t)gsrc;
  tdm_u32x4 g0;
  g0.x = 1u;                                              // count=1, user mode
  g0.y = lds_addr;                                        // LDS byte address
  g0.z = (unsigned)(ga & 0xffffffffull);                  // global_addr[31:0]
  g0.w = (unsigned)((ga >> 32) & 0x1ffffffull) | (2u << 30);  // addr[56:32], type=2
  tdm_i32x8 g1;
  g1.s0 = (int)(1u << 16);                                // data_size=1 (2B), wg_mask=0
  g1.s1 = (int)((dim0 & 0xffffu) << 16);                  // tensor_dim0[15:0]
  g1.s2 = (int)(((dim0 >> 16) & 0xffffu) | ((dim1 & 0xffffu) << 16));
  g1.s3 = (int)(((dim1 >> 16) & 0xffffu) | ((tile0 & 0xffffu) << 16));
  g1.s4 = (int)(tile1 & 0xffffu);                         // tile_dim1
  g1.s5 = (int)(stride0 & 0xffffffffull);                 // tensor_dim0_stride
  g1.s6 = (int)((stride0 >> 32) & 0xffffull);
  g1.s7 = 0;
  tdm_i32x4 z4 = {0, 0, 0, 0};
  tdm_i32x8 z8 = {0, 0, 0, 0, 0, 0, 0, 0};
  __builtin_amdgcn_tensor_load_to_lds(g0, g1, z4, z4, z8, 0);
}

// Issue one 128x32 A-tile + 128x32 B-tile pair (row stride = lda elements).
__device__ __forceinline__ void tdm_stage_pair(bf16* Ad, bf16* Bd,
                                               const bf16* gA, const bf16* gB,
                                               unsigned lda) {
  tdm_load_tile_2d((unsigned)(size_t)(void*)Ad, (const void*)gA, 32u, 128u,
                   lda, 128u, (unsigned long long)lda);
  tdm_load_tile_2d((unsigned)(size_t)(void*)Bd, (const void*)gB, 32u, 128u,
                   lda, 128u, (unsigned long long)lda);
}
#else
#define HAS_TDM 0
#endif

// ---------------- WMMA fragment loaders (contiguous k = 16*(lane/16) + s in A and B) ----------------

// A fragment 16x32 row-major: (r,k) at p[r*ld+k]
__device__ __forceinline__ v16bf load_frag_a(const bf16* p, int ld) {
  int lane = threadIdx.x & 31;
  int r  = lane & 15;
  int kb = (lane & 16) ? 16 : 0;
  const bf16* row = p + r * ld + kb;
  v16bf f;
#pragma unroll
  for (int i = 0; i < 8; ++i) f[i] = row[i];
#pragma unroll
  for (int i = 0; i < 8; ++i) f[8 + i] = row[8 + i];
  return f;
}

// A fragment gathered transposed: logical (r,k) at p[k*ld + r]
__device__ __forceinline__ v16bf load_frag_a_tr(const bf16* p, int ld) {
  int lane = threadIdx.x & 31;
  int r  = lane & 15;
  int kb = (lane & 16) ? 16 : 0;
  v16bf f;
#pragma unroll
  for (int i = 0; i < 8; ++i) f[i] = p[(kb + i) * ld + r];
#pragma unroll
  for (int i = 0; i < 8; ++i) f[8 + i] = p[(kb + 8 + i) * ld + r];
  return f;
}

// B fragment 32x16 from [K][N] row-major: (k,c) at p[k*ld + c]
__device__ __forceinline__ v16bf load_frag_b(const bf16* p, int ld) {
  int lane = threadIdx.x & 31;
  int c  = lane & 15;
  int kb = (lane & 16) ? 16 : 0;
  v16bf f;
#pragma unroll
  for (int i = 0; i < 16; ++i) f[i] = p[(kb + i) * ld + c];
  return f;
}

// B fragment from pre-transposed storage [N][K]: (k,c) at p[c*ld + k]
__device__ __forceinline__ v16bf load_frag_bt(const bf16* p, int ld) {
  int lane = threadIdx.x & 31;
  int c  = lane & 15;
  int kb = (lane & 16) ? 16 : 0;
  const bf16* row = p + c * ld + kb;
  v16bf f;
#pragma unroll
  for (int i = 0; i < 16; ++i) f[i] = row[i];
  return f;
}

__device__ __forceinline__ float fast_sigmoid(float x) {
#if __has_builtin(__builtin_amdgcn_rcpf)
  return __builtin_amdgcn_rcpf(1.0f + __expf(-x));
#else
  return 1.0f / (1.0f + __expf(-x));
#endif
}

// ---------------- conversion kernels ----------------

__global__ __launch_bounds__(256) void k_cvt_bf16(const float* __restrict__ src,
                                                  bf16* __restrict__ dst, int n) {
  for (int i = blockIdx.x * 256 + threadIdx.x; i < n; i += gridDim.x * 256)
    dst[i] = (bf16)src[i];
}

__global__ __launch_bounds__(256) void k_prep_w(const float* __restrict__ Wv,
                                                const float* __restrict__ Wk,
                                                const float* __restrict__ Wq,
                                                const float* __restrict__ Wo,
                                                bf16* __restrict__ WcatT,
                                                bf16* __restrict__ WoT) {
  const int NC = 3072 * 1024;
  const int TOT = NC + 1024 * 1024;
  for (int i = blockIdx.x * 256 + threadIdx.x; i < TOT; i += gridDim.x * 256) {
    if (i < NC) {
      int n = i >> 10, e = i & 1023;
      int sel = n >> 10, cc = n & 1023;
      const float* W = (sel == 0) ? Wv : ((sel == 1) ? Wk : Wq);
      WcatT[i] = (bf16)W[e * 1024 + cc];
    } else {
      int j = i - NC;
      int n = j >> 10, dv = j & 1023;
      WoT[j] = (bf16)Wo[dv * 1024 + n];
    }
  }
}

// ---------------- double-buffered 128x128 GEMM tile core ----------------
// A: [M][K] row-major (ld=lda), Bt: [N][K] row-major (ld=lda).
// As/Bs each hold TWO 128x32 buffers (128*64 elems). 8 waves = 4(m) x 2(n).
template <int KDIM>
__device__ __forceinline__ void gemm_tile_128x128(const bf16* __restrict__ A,
                                                  const bf16* __restrict__ Bt,
                                                  int lda, v8f acc[8],
                                                  bf16* As, bf16* Bs) {
  int tid  = threadIdx.x;
  int wave = tid >> 5;
  int wm   = wave & 3;
  int wn   = wave >> 2;
  int p = 0;
#if HAS_TDM
  if (tid < 32) tdm_stage_pair(As, Bs, A, Bt, (unsigned)lda);
  for (int k0 = 0; k0 < KDIM; k0 += 32) {
    bf16* Ac = As + (p ? 4096 : 0);
    bf16* Bc = Bs + (p ? 4096 : 0);
    if (tid < 32) {
      if (k0 + 32 < KDIM) {
        tdm_stage_pair(As + (p ? 0 : 4096), Bs + (p ? 0 : 4096),
                       A + k0 + 32, Bt + k0 + 32, (unsigned)lda);
        __builtin_amdgcn_s_wait_tensorcnt(2);   // current chunk landed
      } else {
        __builtin_amdgcn_s_wait_tensorcnt(0);
      }
    }
    __syncthreads();
    v16bf af[2], bfrag[4];
#pragma unroll
    for (int mt = 0; mt < 2; ++mt) af[mt] = load_frag_a(Ac + (wm * 32 + mt * 16) * 32, 32);
#pragma unroll
    for (int nt = 0; nt < 4; ++nt) bfrag[nt] = load_frag_bt(Bc + (wn * 64 + nt * 16) * 32, 32);
#pragma unroll
    for (int mt = 0; mt < 2; ++mt)
#pragma unroll
      for (int nt = 0; nt < 4; ++nt)
        acc[mt * 4 + nt] = wmma_bf16(af[mt], bfrag[nt], acc[mt * 4 + nt]);
    __syncthreads();
    p ^= 1;
  }
#else
  int r  = tid >> 1;
  int ko = (tid & 1) * 16;
  const bf16* gA = A + r * lda + ko;
  const bf16* gB = Bt + r * lda + ko;
  uint4 ra0 = ((const uint4*)gA)[0], ra1 = ((const uint4*)gA)[1];
  uint4 rb0 = ((const uint4*)gB)[0], rb1 = ((const uint4*)gB)[1];
  ((uint4*)(As + r * 32 + ko))[0] = ra0; ((uint4*)(As + r * 32 + ko))[1] = ra1;
  ((uint4*)(Bs + r * 32 + ko))[0] = rb0; ((uint4*)(Bs + r * 32 + ko))[1] = rb1;
  for (int k0 = 0; k0 < KDIM; k0 += 32) {
    __syncthreads();
    bf16* Ac = As + (p ? 4096 : 0);
    bf16* Bc = Bs + (p ? 4096 : 0);
    bool more = (k0 + 32 < KDIM);
    if (more) {
      ra0 = ((const uint4*)(gA + k0 + 32))[0]; ra1 = ((const uint4*)(gA + k0 + 32))[1];
      rb0 = ((const uint4*)(gB + k0 + 32))[0]; rb1 = ((const uint4*)(gB + k0 + 32))[1];
      if (k0 + 64 < KDIM) {
        __builtin_prefetch((const void*)(gA + k0 + 64), 0, 0);
        __builtin_prefetch((const void*)(gB + k0 + 64), 0, 0);
      }
    }
    v16bf af[2], bfrag[4];
#pragma unroll
    for (int mt = 0; mt < 2; ++mt) af[mt] = load_frag_a(Ac + (wm * 32 + mt * 16) * 32, 32);
#pragma unroll
    for (int nt = 0; nt < 4; ++nt) bfrag[nt] = load_frag_bt(Bc + (wn * 64 + nt * 16) * 32, 32);
#pragma unroll
    for (int mt = 0; mt < 2; ++mt)
#pragma unroll
      for (int nt = 0; nt < 4; ++nt)
        acc[mt * 4 + nt] = wmma_bf16(af[mt], bfrag[nt], acc[mt * 4 + nt]);
    if (more) {
      bf16* An = As + (p ? 0 : 4096);
      bf16* Bn = Bs + (p ? 0 : 4096);
      ((uint4*)(An + r * 32 + ko))[0] = ra0; ((uint4*)(An + r * 32 + ko))[1] = ra1;
      ((uint4*)(Bn + r * 32 + ko))[0] = rb0; ((uint4*)(Bn + r * 32 + ko))[1] = rb1;
    }
    p ^= 1;
  }
#endif
}

// ---------------- fused qkv projection ----------------

__global__ __launch_bounds__(256) void k_gemm_qkv(const bf16* __restrict__ Xb,
                                                  const bf16* __restrict__ WcatT,
                                                  const float* __restrict__ bk,
                                                  const float* __restrict__ bq,
                                                  bf16* __restrict__ Vb,
                                                  bf16* __restrict__ Kb,
                                                  bf16* __restrict__ Qb) {
  __shared__ bf16 As[128 * 64];   // double-buffered
  __shared__ bf16 Bs[128 * 64];
  int row0 = blockIdx.y * 128;
  int col0 = blockIdx.x * 128;
  v8f acc[8] = {};
  gemm_tile_128x128<1024>(Xb + (size_t)row0 * 1024, WcatT + (size_t)col0 * 1024, 1024, acc, As, Bs);
  int wave = threadIdx.x >> 5, lane = threadIdx.x & 31;
  int wm = wave & 3, wn = wave >> 2;
  int cN = lane & 15, mb = (lane & 16) ? 8 : 0;
#pragma unroll
  for (int mt = 0; mt < 2; ++mt)
#pragma unroll
    for (int nt = 0; nt < 4; ++nt) {
      int col = col0 + wn * 64 + nt * 16 + cN;
      int sel = col >> 10, cc = col & 1023;
#pragma unroll
      for (int i = 0; i < 8; ++i) {
        int row = row0 + wm * 32 + mt * 16 + mb + i;
        float vv = acc[mt * 4 + nt][i];
        if (sel == 0) {
          Vb[(size_t)row * 1024 + cc] = (bf16)(vv * (1.0f - DECAYF));
        } else if (sel == 1) {
          Kb[(size_t)row * 1024 + cc] = (bf16)fast_sigmoid(vv + bk[cc]);
        } else {
          Qb[(size_t)row * 1024 + cc] = (bf16)fast_sigmoid(vv + bq[cc]);
        }
      }
    }
}

// ---------------- intra-block decayed attention ----------------
// grid.x = B*N_blk (128). Stage1: S1=q@k^T (decay masked). Stage2: y=S1@v.
__global__ __launch_bounds__(256) void k_intra(const bf16* __restrict__ Qb,
                                               const bf16* __restrict__ Kb,
                                               const bf16* __restrict__ Vb,
                                               float* __restrict__ Y) {
  __shared__ bf16  SA[128 * 128];   // stage1: 4 staging buffers; stage2: v tile
  __shared__ bf16  Sl[128 * 128];   // bf16 masked attention matrix
  __shared__ float dtab[128];       // decay^j
  bf16* As0 = SA;
  bf16* Bs0 = SA + 4096;
  bf16* As1 = SA + 8192;
  bf16* Bs1 = SA + 12288;
  int bn = blockIdx.x;
  const bf16* q = Qb + (size_t)bn * 128 * 1024;
  const bf16* k = Kb + (size_t)bn * 128 * 1024;
  const bf16* v = Vb + (size_t)bn * 128 * 1024;
  int tid = threadIdx.x, wave = tid >> 5, lane = tid & 31;
  int cN = lane & 15, mb = (lane & 16) ? 8 : 0;
  if (tid < 128) dtab[tid] = __expf((float)tid * LOG_DECAY);

  // Stage 1: S1 = q @ k^T  (128x128, K=1024), 2-deep TDM pipeline
  v8f acc[8] = {};
  int p = 0;
#if HAS_TDM
  if (tid < 32) tdm_stage_pair(As0, Bs0, q, k, 1024u);
#endif
  for (int k0 = 0; k0 < 1024; k0 += 32) {
    bf16* Ac = p ? As1 : As0;
    bf16* Bc = p ? Bs1 : Bs0;
#if HAS_TDM
    if (tid < 32) {
      if (k0 + 32 < 1024) {
        tdm_stage_pair(p ? As0 : As1, p ? Bs0 : Bs1, q + k0 + 32, k + k0 + 32, 1024u);
        __builtin_amdgcn_s_wait_tensorcnt(2);
      } else {
        __builtin_amdgcn_s_wait_tensorcnt(0);
      }
    }
#else
    {
      int r = tid >> 1, ko = (tid & 1) * 16;
      ((uint4*)(Ac + r * 32 + ko))[0] = ((const uint4*)(q + r * 1024 + k0 + ko))[0];
      ((uint4*)(Ac + r * 32 + ko))[1] = ((const uint4*)(q + r * 1024 + k0 + ko))[1];
      ((uint4*)(Bc + r * 32 + ko))[0] = ((const uint4*)(k + r * 1024 + k0 + ko))[0];
      ((uint4*)(Bc + r * 32 + ko))[1] = ((const uint4*)(k + r * 1024 + k0 + ko))[1];
    }
#endif
    __syncthreads();
    v16bf af = load_frag_a(Ac + (wave * 16) * 32, 32);
#pragma unroll
    for (int nt = 0; nt < 8; ++nt) {
      v16bf bfrag = load_frag_bt(Bc + (nt * 16) * 32, 32);
      acc[nt] = wmma_bf16(af, bfrag, acc[nt]);
    }
    __syncthreads();
    p ^= 1;
  }
  // decay mask -> Sl (bf16)
#pragma unroll
  for (int nt = 0; nt < 8; ++nt)
#pragma unroll
    for (int i = 0; i < 8; ++i) {
      int t = wave * 16 + mb + i, s = nt * 16 + cN;
      float f = (t >= s) ? dtab[t - s] : 0.0f;
      Sl[t * 128 + s] = (bf16)(acc[nt][i] * f);
    }
  __syncthreads();

  // Stage 2: y = Sl @ v  (K=128), v staged 64 columns at a time in LDS
  for (int c0 = 0; c0 < 1024; c0 += 64) {
    {
      int r = tid >> 1, off = (tid & 1) * 32;
      const uint4* gv = (const uint4*)(v + r * 1024 + c0 + off);
      uint4* lv = (uint4*)(SA + r * 64 + off);
      lv[0] = gv[0]; lv[1] = gv[1]; lv[2] = gv[2]; lv[3] = gv[3];
    }
    __syncthreads();
#pragma unroll
    for (int vt = 0; vt < 4; ++vt) {
      v8f yacc = {};
#pragma unroll
      for (int kc = 0; kc < 4; ++kc) {
        v16bf af = load_frag_a(Sl + (wave * 16) * 128 + kc * 32, 128);
        v16bf bfrag = load_frag_b(SA + (kc * 32) * 64 + vt * 16, 64);
        yacc = wmma_bf16(af, bfrag, yacc);
      }
#pragma unroll
      for (int i = 0; i < 8; ++i) {
        int t = wave * 16 + mb + i;
        Y[((size_t)bn * 128 + t) * 1024 + c0 + vt * 16 + cN] = yacc[i];
      }
    }
    __syncthreads();
  }
}

// ---------------- cross-block state recurrence ----------------
// grid = (Dk/64, Dv/64, B). WG owns a 64x64 state tile across the 32-step scan.
__global__ __launch_bounds__(256) void k_state(const bf16* __restrict__ Qb,
                                               const bf16* __restrict__ Kb,
                                               const bf16* __restrict__ Vb,
                                               float* __restrict__ Y) {
  __shared__ float Sf[64 * 64];
  __shared__ bf16  Sb[64 * 64];
  __shared__ bf16  Kt[32 * 64];     // k chunk [t=32][d=64]
  __shared__ bf16  Vt[32 * 64];     // weighted v chunk [t=32][v=64]
  __shared__ float dtab[128];       // decay^t
  int b = blockIdx.z, d0 = blockIdx.x * 64, v0 = blockIdx.y * 64;
  int tid = threadIdx.x, wave = tid >> 5, lane = tid & 31;
  int cN = lane & 15, mb = (lane & 16) ? 8 : 0;
  for (int i = tid; i < 64 * 64; i += 256) { Sf[i] = 0.0f; Sb[i] = (bf16)0.0f; }
  if (tid < 128) dtab[tid] = __expf((float)tid * LOG_DECAY);
  __syncthreads();
  const float BD = __expf(128.0f * LOG_DECAY);  // decay^T_BLK

  for (int n = 0; n < 32; ++n) {
    size_t base = ((size_t)(b * 32 + n)) * 128 * 1024;
    const bf16* q = Qb + base;
    const bf16* k = Kb + base;
    const bf16* v = Vb + base;

    // y2 partial: q[:, d0:d0+64] @ S (K=64)
    if (n > 0) {
      v8f p[4] = {};
#pragma unroll
      for (int kc = 0; kc < 2; ++kc) {
        v16bf af = load_frag_a(q + (wave * 16) * 1024 + d0 + kc * 32, 1024);
#pragma unroll
        for (int vt = 0; vt < 4; ++vt) {
          v16bf bfrag = load_frag_b(Sb + (kc * 32) * 64 + vt * 16, 64);
          p[vt] = wmma_bf16(af, bfrag, p[vt]);
        }
      }
#pragma unroll
      for (int vt = 0; vt < 4; ++vt)
#pragma unroll
        for (int i = 0; i < 8; ++i) {
          int t = wave * 16 + mb + i;
          atomicAdd(&Y[((size_t)(b * 32 + n) * 128 + t) * 1024 + v0 + vt * 16 + cN],
                    p[vt][i] * dtab[t]);
        }
    }

    // kv_blk tile: k^T @ (v * decay^(127-t)), K=128 in 4 chunks staged via LDS
    int md = (wave & 3) * 16;
    int ch = (wave >> 2) * 32;
    v8f kv[2] = {};
#pragma unroll 1
    for (int kc = 0; kc < 4; ++kc) {
      __syncthreads();   // prior-iteration readers (incl. y2 reads of Sb) done
      {
        int r = tid >> 3, off = (tid & 7) * 8;   // 32 rows x 64 cols, 8 elems/thread
        int t = kc * 32 + r;
        ((uint4*)(Kt + r * 64 + off))[0] = ((const uint4*)(k + t * 1024 + d0 + off))[0];
        union { uint4 u; bf16 h[8]; } raw;
        raw.u = ((const uint4*)(v + t * 1024 + v0 + off))[0];
        float w = dtab[127 - t];
#pragma unroll
        for (int e = 0; e < 8; ++e) raw.h[e] = (bf16)((float)raw.h[e] * w);
        ((uint4*)(Vt + r * 64 + off))[0] = raw.u;
      }
      __syncthreads();
      v16bf af = load_frag_a_tr(Kt + md, 64);
#pragma unroll
      for (int vt = 0; vt < 2; ++vt) {
        v16bf bfrag = load_frag_b(Vt + ch + vt * 16, 64);
        kv[vt] = wmma_bf16(af, bfrag, kv[vt]);
      }
    }
    __syncthreads();
    // S = BD*S + DECAY*kv_blk   (wave-disjoint regions)
#pragma unroll
    for (int vt = 0; vt < 2; ++vt)
#pragma unroll
      for (int i = 0; i < 8; ++i) {
        int dl = md + mb + i;
        int vl = ch + vt * 16 + cN;
        float s = BD * Sf[dl * 64 + vl] + DECAYF * kv[vt][i];
        Sf[dl * 64 + vl] = s;
        Sb[dl * 64 + vl] = (bf16)s;
      }
    __syncthreads();
  }
}

// ---------------- output projection ----------------

__global__ __launch_bounds__(256) void k_gemm_out(const bf16* __restrict__ Ybf,
                                                  const bf16* __restrict__ WoT,
                                                  float* __restrict__ out) {
  __shared__ bf16 As[128 * 64];   // double-buffered
  __shared__ bf16 Bs[128 * 64];
  int row0 = blockIdx.y * 128;
  int col0 = blockIdx.x * 128;
  v8f acc[8] = {};
  gemm_tile_128x128<1024>(Ybf + (size_t)row0 * 1024, WoT + (size_t)col0 * 1024, 1024, acc, As, Bs);
  int wave = threadIdx.x >> 5, lane = threadIdx.x & 31;
  int wm = wave & 3, wn = wave >> 2;
  int cN = lane & 15, mb = (lane & 16) ? 8 : 0;
#pragma unroll
  for (int mt = 0; mt < 2; ++mt)
#pragma unroll
    for (int nt = 0; nt < 4; ++nt)
#pragma unroll
      for (int i = 0; i < 8; ++i) {
        int row = row0 + wm * 32 + mt * 16 + mb + i;
        int col = col0 + wn * 64 + nt * 16 + cN;
        out[(size_t)row * 1024 + col] = acc[mt * 4 + nt][i];
      }
}

// ---------------- host launch ----------------

extern "C" void kernel_launch(void* const* d_in, const int* in_sizes, int n_in,
                              void* d_out, int out_size, void* d_ws, size_t ws_size,
                              hipStream_t stream) {
  const float* x  = (const float*)d_in[0];
  const float* Wv = (const float*)d_in[1];
  const float* Wk = (const float*)d_in[2];
  const float* bk = (const float*)d_in[3];
  const float* Wq = (const float*)d_in[4];
  const float* bq = (const float*)d_in[5];
  const float* Wo = (const float*)d_in[6];
  float* out = (float*)d_out;

  char* ws = (char*)d_ws;
  bf16* Xb    = (bf16*)(ws);                       // 32 MB (reused as Ybf)
  bf16* WcatT = (bf16*)(ws + (32ull << 20));       //  6 MB
  bf16* WoT   = (bf16*)(ws + (38ull << 20));       //  2 MB
  bf16* Vb    = (bf16*)(ws + (40ull << 20));       // 32 MB
  bf16* Kb    = (bf16*)(ws + (72ull << 20));       // 32 MB
  bf16* Qb    = (bf16*)(ws + (104ull << 20));      // 32 MB
  float* Y    = (float*)(ws + (136ull << 20));     // 64 MB
  bf16* Ybf   = Xb;

  const int MT = 16384;  // B*T

  k_cvt_bf16<<<dim3(2048), dim3(256), 0, stream>>>(x, Xb, MT * 1024);
  k_prep_w<<<dim3(2048), dim3(256), 0, stream>>>(Wv, Wk, Wq, Wo, WcatT, WoT);
  k_gemm_qkv<<<dim3(24, 128), dim3(256), 0, stream>>>(Xb, WcatT, bk, bq, Vb, Kb, Qb);
  k_intra<<<dim3(128), dim3(256), 0, stream>>>(Qb, Kb, Vb, Y);
  k_state<<<dim3(16, 16, 4), dim3(256), 0, stream>>>(Qb, Kb, Vb, Y);
  k_cvt_bf16<<<dim3(2048), dim3(256), 0, stream>>>(Y, Ybf, MT * 1024);
  k_gemm_out<<<dim3(8, 128), dim3(256), 0, stream>>>(Ybf, WoT, out);
}
